// MultiHeadAttention_49374944035451
// MI455X (gfx1250) — matrix operand order repaired and checked by reading, not compiled, not run on hardware
//
#include <hip/hip_runtime.h>
#include <hip/hip_bf16.h>

// The Python harness owns main(), I/O, device malloc/memcpy, and timing.
// This file implements kernel_launch() and the kernels it needs.
// All kernels launch on `stream`; d_ws is used as scratch (~64 MB needed).

typedef __bf16 bf16;
typedef __attribute__((ext_vector_type(16))) __bf16 v16bf;
typedef __attribute__((ext_vector_type(8)))  __bf16 v8bf;
typedef __attribute__((ext_vector_type(8)))  float   v8f;
typedef int v4i32 __attribute__((vector_size(16)));   // matches builtin param type

#define DM    1024
#define NH    16
#define DKH   64
#define SEQL  2048
#define BATCH 2
#define MTOT  (BATCH * SEQL)   // 4096 rows in the "token" GEMMs

// ---------------------------------------------------------------------------
// Async global -> LDS staging (gfx1250 GLOBAL_LOAD_ASYNC_TO_LDS_B128, ASYNCcnt)
// with a VGPR-bounce fallback if the toolchain lacks the builtins.
// ---------------------------------------------------------------------------
#if defined(__has_builtin)
#if __has_builtin(__builtin_amdgcn_global_load_async_to_lds_b128) && \
    __has_builtin(__builtin_amdgcn_s_wait_asynccnt)
#define USE_ASYNC_G2L 1
#endif
#endif
#ifndef USE_ASYNC_G2L
#define USE_ASYNC_G2L 0
#endif

__device__ __forceinline__ void copy16_g2l(const bf16* g, bf16* l) {
#if USE_ASYNC_G2L
  // Generic LDS address low 32 bits == LDS offset (aperture rule), so the
  // truncating cast yields the HW-required AS(3) pointer value.
  __builtin_amdgcn_global_load_async_to_lds_b128(
      (__attribute__((address_space(1))) v4i32*)(unsigned long long)g,
      (__attribute__((address_space(3))) v4i32*)(unsigned)(unsigned long long)l,
      0, 0);
#else
  *(v8bf*)l = *(const v8bf*)g;
#endif
}
__device__ __forceinline__ void wait_g2l() {
#if USE_ASYNC_G2L
  __builtin_amdgcn_s_wait_asynccnt(0);   // must precede the workgroup barrier
#endif
}

// ---------------------------------------------------------------------------
// WMMA fragment loaders (bf16, 16x16x32).
// A (16x32, lane=row m=lane&15): elems [0..7] = K kb..kb+7, [8..15] = K kb+16..kb+23,
//   kb = (lane<16 ? 0 : 8).
// B (32x16, lane=col n=lane&15): elems [0..7] = K kb..kb+7, [8..15] = K kb+16..kb+23, kb = (lane<16 ? 0 : 8).
// C/D (16x16 f32): vgpr v -> row v + (lane<16?0:8), col = lane&15.
// ---------------------------------------------------------------------------
__device__ __forceinline__ v16bf ld_afrag(const bf16* p) {
  v8bf lo = *(const v8bf*)p;
  v8bf hi = *(const v8bf*)(p + 16);
  v16bf r;
#pragma unroll
  for (int e = 0; e < 8; ++e) { r[e] = lo[e]; r[e + 8] = hi[e]; }
  return r;
}
__device__ __forceinline__ v16bf ld_bfrag(const bf16* p) {
  v8bf lo = *(const v8bf*)p;
  v8bf hi = *(const v8bf*)(p + 16);
  v16bf r;
#pragma unroll
  for (int e = 0; e < 8; ++e) { r[e] = lo[e]; r[e + 8] = hi[e]; }
  return r;
}

// ---------------------------------------------------------------------------
// fp32 -> bf16 conversion
// ---------------------------------------------------------------------------
__global__ void cvt_f32_bf16(const float* __restrict__ s, bf16* __restrict__ d, int n) {
  int i = blockIdx.x * blockDim.x + threadIdx.x;
  int stride = gridDim.x * blockDim.x;
  for (; i < n; i += stride) d[i] = (bf16)s[i];
}

// ---------------------------------------------------------------------------
// Tiled WMMA GEMM: C[m,n] = sum_k A[m,k] * W[n,k] + bias[n]   (i.e. A @ W^T + b)
// Block tile 128x128, 8 waves, wave tile 32x64 (head-aligned on N).
// MODE 0: store fp32 to outF            (final output projection)
// MODE 1: RoPE epilogue, store bf16 to outB laid out [B,H,L,64]   (Q, K)
// MODE 2: store bf16 transposed to outB laid out [B,H,64,L]       (V)
// ---------------------------------------------------------------------------
template <int MODE>
__global__ __launch_bounds__(256) void gemm_wmma(
    const bf16* __restrict__ A, const bf16* __restrict__ W,
    const float* __restrict__ bias, float* __restrict__ outF,
    bf16* __restrict__ outB) {
  constexpr int BM = 128, BN = 128, BK = 64, K = DM, N = DM;
  __shared__ __align__(16) bf16 sA[BM][BK];
  __shared__ __align__(16) bf16 sB[BN][BK];
  const int tid = threadIdx.x, lane = tid & 31, wave = tid >> 5;
  const int wm = wave & 3, wn = wave >> 2;
  const int m0 = (blockIdx.x / (N / BN)) * BM;
  const int n0 = (blockIdx.x % (N / BN)) * BN;
  const int c = lane & 15;
  const int hh = (lane & 16) ? 1 : 0;   // half-wave selector

  v8f acc[2][4] = {};

  for (int k0 = 0; k0 < K; k0 += BK) {
#pragma unroll
    for (int i = 0; i < 4; ++i) {
      int idx = tid + i * 256;
      int r = idx >> 3, cc = (idx & 7) * 8;
      copy16_g2l(A + (size_t)(m0 + r) * K + k0 + cc, &sA[r][cc]);
      copy16_g2l(W + (size_t)(n0 + r) * K + k0 + cc, &sB[r][cc]);
    }
    if (k0 + BK < K) {  // hint next tiles into cache (global_prefetch_b8)
      __builtin_prefetch((const void*)(A + (size_t)(m0 + (tid >> 1)) * K + k0 + BK), 0, 1);
      __builtin_prefetch((const void*)(W + (size_t)(n0 + (tid >> 1)) * K + k0 + BK), 0, 1);
    }
    wait_g2l();
    __syncthreads();
#pragma unroll
    for (int kk = 0; kk < 2; ++kk) {
      v16bf af[2], wf[4];
#pragma unroll
      for (int mf = 0; mf < 2; ++mf)
        af[mf] = ld_afrag(&sA[wm * 32 + mf * 16 + c][kk * 32 + hh * 8]);
#pragma unroll
      for (int nf = 0; nf < 4; ++nf)
        wf[nf] = ld_bfrag(&sB[wn * 64 + nf * 16 + c][kk * 32 + hh * 8]);
#pragma unroll
      for (int mf = 0; mf < 2; ++mf)
#pragma unroll
        for (int nf = 0; nf < 4; ++nf)
          acc[mf][nf] = __builtin_amdgcn_wmma_f32_16x16x32_bf16(
              false, af[mf], false, wf[nf], (short)0, acc[mf][nf], false, false);
    }
    __syncthreads();
  }

  const int mbase = m0 + wm * 32;
  const int nbase = n0 + wn * 64;

  if (MODE == 0) {
#pragma unroll
    for (int mf = 0; mf < 2; ++mf)
#pragma unroll
      for (int nf = 0; nf < 4; ++nf)
#pragma unroll
        for (int v = 0; v < 8; ++v) {
          int m = mbase + mf * 16 + v + hh * 8;
          int n = nbase + nf * 16 + c;
          outF[(size_t)m * N + n] = acc[mf][nf][v] + bias[n];
        }
  } else if (MODE == 1) {
    const int h = nbase >> 6;  // wave N-tile is one full head
#pragma unroll
    for (int mf = 0; mf < 2; ++mf)
#pragma unroll
      for (int nf = 0; nf < 2; ++nf)
#pragma unroll
        for (int v = 0; v < 8; ++v) {
          int m = mbase + mf * 16 + v + hh * 8;
          int b = m >> 11, l = m & (SEQL - 1);
          int d = nf * 16 + c;  // 0..31
          float x1 = acc[mf][nf][v]     + bias[nbase + d];
          float x2 = acc[mf][nf + 2][v] + bias[nbase + d + 32];
          float ang = (float)l * __powf(10000.f, -(float)d * (1.f / 32.f));
          float cs = __cosf(ang), sn = __sinf(ang);
          bf16* dst = outB + ((size_t)(b * NH + h) * SEQL + l) * DKH;
          dst[d]      = (bf16)(x1 * cs - x2 * sn);
          dst[d + 32] = (bf16)(x2 * cs + x1 * sn);
        }
  } else {  // MODE 2: V transposed [B,H,64,L]
    const int h = nbase >> 6;
#pragma unroll
    for (int mf = 0; mf < 2; ++mf)
#pragma unroll
      for (int nf = 0; nf < 4; ++nf)
#pragma unroll
        for (int v = 0; v < 8; ++v) {
          int m = mbase + mf * 16 + v + hh * 8;
          int b = m >> 11, l = m & (SEQL - 1);
          int d = nf * 16 + c;
          float x = acc[mf][nf][v] + bias[nbase + d];
          outB[((size_t)(b * NH + h) * DKH + d) * SEQL + l] = (bf16)x;
        }
  }
}

// ---------------------------------------------------------------------------
// Flash attention: block = (b,h, 256 q-rows); 8 waves x 32 q-rows each.
// K/V^T tiles (32 keys) staged in LDS (async); S = Q K^T and O += P V via
// bf16 WMMA; online softmax with 16-lane shfl_xor butterflies (wave32).
// Output ctx stored bf16 as [B*L, 1024] (feature = h*64 + d) for out-proj.
// ---------------------------------------------------------------------------
__global__ __launch_bounds__(256) void flash_attn(
    const bf16* __restrict__ Qh, const bf16* __restrict__ Kh,
    const bf16* __restrict__ VT, bf16* __restrict__ ctx) {
  __shared__ __align__(16) bf16 sK[32][DKH];       // [key][d]
  __shared__ __align__(16) bf16 sVT[DKH][32];      // [d][key]
  __shared__ __align__(16) bf16 sP[8][32][32];     // per-wave P tile [q][key]
  const int tid = threadIdx.x, lane = tid & 31, wave = tid >> 5;
  const int c = lane & 15;
  const int hh = (lane & 16) ? 1 : 0;
  const int bh = blockIdx.x >> 3;   // 0..31  (b*16 + h)
  const int qblk = blockIdx.x & 7;
  const int b = bh >> 4, h = bh & 15;
  const int qbase = qblk * 256 + wave * 32;
  const bf16* Qb = Qh + (size_t)bh * SEQL * DKH;
  const bf16* Kb = Kh + (size_t)bh * SEQL * DKH;
  const bf16* Vb = VT + (size_t)bh * DKH * SEQL;

  // Preload Q fragments for this wave's 32 rows (2 mfrag x 2 K-chunks).
  v16bf qf[2][2];
#pragma unroll
  for (int mf = 0; mf < 2; ++mf)
#pragma unroll
    for (int kk = 0; kk < 2; ++kk)
      qf[mf][kk] = ld_afrag(Qb + (size_t)(qbase + mf * 16 + c) * DKH + kk * 32 + hh * 8);

  v8f o[2][4] = {};
  float mrow[2][8], lrow[2][8];
#pragma unroll
  for (int mf = 0; mf < 2; ++mf)
#pragma unroll
    for (int v = 0; v < 8; ++v) { mrow[mf][v] = -3.0e38f; lrow[mf][v] = 0.f; }

  for (int k0 = 0; k0 < SEQL; k0 += 32) {
    {
      int key = tid >> 3, d8 = (tid & 7) * 8;
      copy16_g2l(Kb + (size_t)(k0 + key) * DKH + d8, &sK[key][d8]);
      int dv = tid >> 2, kq8 = (tid & 3) * 8;
      copy16_g2l(Vb + (size_t)dv * SEQL + k0 + kq8, &sVT[dv][kq8]);
    }
    wait_g2l();
    __syncthreads();

    // S = Q K^T  (32q x 32keys per wave)
    v8f s[2][2] = {};
#pragma unroll
    for (int kk = 0; kk < 2; ++kk) {
      v16bf kf[2];
#pragma unroll
      for (int nf = 0; nf < 2; ++nf)
        kf[nf] = ld_bfrag(&sK[nf * 16 + c][kk * 32 + hh * 8]);
#pragma unroll
      for (int mf = 0; mf < 2; ++mf)
#pragma unroll
        for (int nf = 0; nf < 2; ++nf)
          s[mf][nf] = __builtin_amdgcn_wmma_f32_16x16x32_bf16(
              false, qf[mf][kk], false, kf[nf], (short)0, s[mf][nf], false, false);
    }

    // Online softmax + write P (bf16) to wave-private LDS tile.
#pragma unroll
    for (int mf = 0; mf < 2; ++mf) {
#pragma unroll
      for (int v = 0; v < 8; ++v) {
        float x0 = s[mf][0][v] * 0.125f;   // / sqrt(64)
        float x1 = s[mf][1][v] * 0.125f;
        float t = fmaxf(x0, x1);
#pragma unroll
        for (int off = 1; off < 16; off <<= 1)
          t = fmaxf(t, __shfl_xor(t, off, 32));
        float mo = mrow[mf][v];
        float mn = fmaxf(mo, t);
        float alpha = __expf(mo - mn);
        mrow[mf][v] = mn;
        float p0 = __expf(x0 - mn), p1 = __expf(x1 - mn);
        float rs = p0 + p1;
#pragma unroll
        for (int off = 1; off < 16; off <<= 1)
          rs += __shfl_xor(rs, off, 32);
        lrow[mf][v] = lrow[mf][v] * alpha + rs;
#pragma unroll
        for (int nf = 0; nf < 4; ++nf) o[mf][nf][v] *= alpha;
        int r = mf * 16 + v + hh * 8;
        sP[wave][r][c]      = (bf16)p0;
        sP[wave][r][c + 16] = (bf16)p1;
      }
    }

    // O += P * V  (K-dim = 32 keys, one WMMA per (mf,nf))
    {
      v16bf pf[2], vf[4];
#pragma unroll
      for (int mf = 0; mf < 2; ++mf)
        pf[mf] = ld_afrag(&sP[wave][mf * 16 + c][hh * 8]);
#pragma unroll
      for (int nf = 0; nf < 4; ++nf)
        vf[nf] = ld_bfrag(&sVT[nf * 16 + c][hh * 8]);
#pragma unroll
      for (int mf = 0; mf < 2; ++mf)
#pragma unroll
        for (int nf = 0; nf < 4; ++nf)
          o[mf][nf] = __builtin_amdgcn_wmma_f32_16x16x32_bf16(
              false, pf[mf], false, vf[nf], (short)0, o[mf][nf], false, false);
    }
    __syncthreads();
  }

  // ctx[token, h*64 + d] = O / l
#pragma unroll
  for (int mf = 0; mf < 2; ++mf)
#pragma unroll
    for (int v = 0; v < 8; ++v) {
      float invl = 1.f / lrow[mf][v];
      int tok = b * SEQL + qbase + mf * 16 + v + hh * 8;
      bf16* dst = ctx + (size_t)tok * DM + h * DKH;
#pragma unroll
      for (int nf = 0; nf < 4; ++nf)
        dst[nf * 16 + c] = (bf16)(o[mf][nf][v] * invl);
    }
}

// ---------------------------------------------------------------------------
// Launcher
// ---------------------------------------------------------------------------
extern "C" void kernel_launch(void* const* d_in, const int* in_sizes, int n_in,
                              void* d_out, int out_size, void* d_ws, size_t ws_size,
                              hipStream_t stream) {
  (void)in_sizes; (void)n_in; (void)out_size; (void)ws_size;
  const float* q  = (const float*)d_in[0];
  const float* k  = (const float*)d_in[1];
  const float* v  = (const float*)d_in[2];
  const float* Wq = (const float*)d_in[3];
  const float* bq = (const float*)d_in[4];
  const float* Wk = (const float*)d_in[5];
  const float* bk = (const float*)d_in[6];
  const float* Wv = (const float*)d_in[7];
  const float* bv = (const float*)d_in[8];
  const float* Wo = (const float*)d_in[9];
  const float* bo = (const float*)d_in[10];
  float* out = (float*)d_out;

  // Workspace layout (bf16 buffers, ~64 MB total)
  char* ws = (char*)d_ws;
  const size_t SZ_ACT = (size_t)MTOT * DM * sizeof(bf16);  // 8 MB
  const size_t SZ_W   = (size_t)DM * DM * sizeof(bf16);    // 2 MB
  bf16* qb  = (bf16*)ws;                 ws += SZ_ACT;
  bf16* kb  = (bf16*)ws;                 ws += SZ_ACT;
  bf16* vb  = (bf16*)ws;                 ws += SZ_ACT;
  bf16* Wqb = (bf16*)ws;                 ws += SZ_W;
  bf16* Wkb = (bf16*)ws;                 ws += SZ_W;
  bf16* Wvb = (bf16*)ws;                 ws += SZ_W;
  bf16* Wob = (bf16*)ws;                 ws += SZ_W;
  bf16* Qhb = (bf16*)ws;                 ws += SZ_ACT;   // [B,H,L,64]
  bf16* Khb = (bf16*)ws;                 ws += SZ_ACT;   // [B,H,L,64]
  bf16* VTb = (bf16*)ws;                 ws += SZ_ACT;   // [B,H,64,L]
  bf16* ctx = (bf16*)ws;                 ws += SZ_ACT;   // [B*L, 1024]

  const int NACT = MTOT * DM;   // 4 M elements
  const int NW   = DM * DM;     // 1 M elements
  cvt_f32_bf16<<<512, 256, 0, stream>>>(q,  qb,  NACT);
  cvt_f32_bf16<<<512, 256, 0, stream>>>(k,  kb,  NACT);
  cvt_f32_bf16<<<512, 256, 0, stream>>>(v,  vb,  NACT);
  cvt_f32_bf16<<<256, 256, 0, stream>>>(Wq, Wqb, NW);
  cvt_f32_bf16<<<256, 256, 0, stream>>>(Wk, Wkb, NW);
  cvt_f32_bf16<<<256, 256, 0, stream>>>(Wv, Wvb, NW);
  cvt_f32_bf16<<<256, 256, 0, stream>>>(Wo, Wob, NW);

  const int gemm_grid = (MTOT / 128) * (DM / 128);  // 256 blocks
  gemm_wmma<1><<<gemm_grid, 256, 0, stream>>>(qb, Wqb, bq, nullptr, Qhb);
  gemm_wmma<1><<<gemm_grid, 256, 0, stream>>>(kb, Wkb, bk, nullptr, Khb);
  gemm_wmma<2><<<gemm_grid, 256, 0, stream>>>(vb, Wvb, bv, nullptr, VTb);

  const int attn_grid = BATCH * NH * (SEQL / 256);  // 256 blocks
  flash_attn<<<attn_grid, 256, 0, stream>>>(Qhb, Khb, VTb, ctx);

  gemm_wmma<0><<<gemm_grid, 256, 0, stream>>>(ctx, Wob, bo, out, nullptr);
}
